// IPMPDenoiser_19705309954759
// MI455X (gfx1250) — matrix operand order repaired and checked by reading, not compiled, not run on hardware
//
#include <hip/hip_runtime.h>
#include <cstdint>
#include <cstddef>

// ---------------------------------------------------------------------------
// Types / helpers
// ---------------------------------------------------------------------------
typedef __attribute__((ext_vector_type(16))) __bf16 v16bf;
typedef __attribute__((ext_vector_type(8)))  float  v8f;

#define DEVFN __device__ __forceinline__

DEVFN unsigned short f2bf(float f) {
  unsigned u = __float_as_uint(f);
  unsigned r = u + 0x7FFFu + ((u >> 16) & 1u);   // round-to-nearest-even
  return (unsigned short)(r >> 16);
}
DEVFN float bf2f(unsigned short h) { return __uint_as_float(((unsigned)h) << 16); }

DEVFN float dot3(const float* a, const float* b) { return a[0]*b[0] + a[1]*b[1] + a[2]*b[2]; }
DEVFN void  cross3(const float* a, const float* b, float* o) {
  o[0] = a[1]*b[2] - a[2]*b[1];
  o[1] = a[2]*b[0] - a[0]*b[2];
  o[2] = a[0]*b[1] - a[1]*b[0];
}
DEVFN void unit3(const float* a, float* o, float eps) {
  float n = sqrtf(a[0]*a[0] + a[1]*a[1] + a[2]*a[2]) + eps;
  o[0] = a[0] / n; o[1] = a[1] / n; o[2] = a[2] / n;
}

// ---------------------------------------------------------------------------
// Weight packing: f32 row-major (K x N) -> bf16 TRANSPOSED padded (Np x Kp).
// Column-major weights make the GEMM B-tile loader fully vectorized
// (uint4 global loads + uint4 LDS stores, no b16 scatter in the hot loop).
// ---------------------------------------------------------------------------
__global__ void pack_w_kernel(const float* w, unsigned short* wp,
                              int K, int N, int Kp, int Np) {
  int idx = blockIdx.x * blockDim.x + threadIdx.x;
  if (idx >= Kp * Np) return;
  int n = idx / Kp, k = idx % Kp;
  float v = (k < K && n < N) ? w[(size_t)k * N + n] : 0.f;
  wp[(size_t)n * Kp + k] = f2bf(v);
}

__global__ void pack_b_kernel(const float* b, float* bp, int N, int Np) {
  int idx = blockIdx.x * blockDim.x + threadIdx.x;
  if (idx >= Np) return;
  bp[idx] = (idx < N) ? b[idx] : 0.f;
}

__global__ void copy_f32_kernel(const float* a, float* b, int n) {
  int i = blockIdx.x * blockDim.x + threadIdx.x;
  if (i < n) b[i] = a[i];
}

// ---------------------------------------------------------------------------
// WMMA GEMM: C[M x Np] = act(A[M x K](bf16) @ W + bias)
//   W passed TRANSPOSED: Wt[Np x K] bf16, k-contiguous.
// Block tile 64x64, 8 waves; wave computes 16(M) x 32(N).
// K staged 64 per iteration (4 WMMAs per barrier pair); register prefetch
// pipeline overlaps the next tile's global loads with the WMMA burst.
// GATHER: edge message-input concat gathered on the fly:
//   k<512: h[e/30], k<1024: h[ei0[e]], k<1152: z[e], else pd[e] (width 64)
// Requires M%64==0, K%64==0, Np%64==0.
// ---------------------------------------------------------------------------
struct TilePkt { uint4 a0, a1, b0, b1; };

template <bool GATHER>
DEVFN TilePkt load_tiles(const unsigned short* __restrict__ A,
                         const unsigned short* __restrict__ Wt,
                         const unsigned short* __restrict__ hb,
                         const unsigned short* __restrict__ zbuf,
                         const unsigned short* __restrict__ pdbuf,
                         int e, int e30, int eg, int K,
                         int k0, int n0, int arow, int aseg) {
  TilePkt p;
  if (!GATHER) {
    const unsigned short* ar = A + (size_t)e * K + k0 + aseg * 8;
    p.a0 = *(const uint4*)ar;
    p.a1 = *(const uint4*)(ar + 32);
  } else {
#pragma unroll
    for (int c = 0; c < 2; ++c) {
      int kg = k0 + aseg * 8 + 32 * c;
      const unsigned short* src;
      if      (kg < 512)  src = hb    + (size_t)e30 * 512 + kg;
      else if (kg < 1024) src = hb    + (size_t)eg  * 512 + (kg - 512);
      else if (kg < 1152) src = zbuf  + (size_t)e * 128 + (kg - 1024);
      else                src = pdbuf + (size_t)e * 64  + (kg - 1152);
      if (c == 0) p.a0 = *(const uint4*)src; else p.a1 = *(const uint4*)src;
    }
  }
  // B: same access shape as A (Wt is column-major, k-contiguous)
  const unsigned short* br = Wt + (size_t)(n0 + arow) * K + k0 + aseg * 8;
  p.b0 = *(const uint4*)br;
  p.b1 = *(const uint4*)(br + 32);
  return p;
}

template <bool GATHER>
__global__ __launch_bounds__(256)
void gemm_wmma(const unsigned short* __restrict__ A,
               const unsigned short* __restrict__ Wt,
               const float* __restrict__ bias,
               float* __restrict__ outF, unsigned short* __restrict__ outH,
               int M, int K, int Np, int relu,
               const unsigned short* __restrict__ hb,
               const unsigned short* __restrict__ zbuf,
               const unsigned short* __restrict__ pdbuf,
               const int* __restrict__ ei0) {
  __shared__ unsigned short lsA[64 * 64];   // [row][k]   8 KB
  __shared__ unsigned short lsB[64 * 64];   // [n][k]     8 KB

  const int tid  = threadIdx.x;
  const int lane = tid & 31;
  const int wv   = tid >> 5;
  const int wm   = wv & 3;          // 4 M-subtiles of 16
  const int wn   = wv >> 2;         // 2 N-subtiles of 32
  const int m0   = blockIdx.x * 64;
  const int n0   = blockIdx.y * 64;

  const int arow = tid >> 2, aseg = tid & 3;   // 64 rows x (2 x 8-col chunks)

  const int e   = m0 + arow;
  const int e30 = GATHER ? (e / 30) : 0;
  const int eg  = GATHER ? ei0[e] : 0;

  v8f acc0 = {}, acc1 = {};

  TilePkt pk = load_tiles<GATHER>(A, Wt, hb, zbuf, pdbuf, e, e30, eg,
                                  K, 0, n0, arow, aseg);

  for (int k0 = 0; k0 < K; k0 += 64) {
    // ---- stage tiles to LDS (all vectorized b128) ----
    *(uint4*)&lsA[arow * 64 + aseg * 8]      = pk.a0;
    *(uint4*)&lsA[arow * 64 + 32 + aseg * 8] = pk.a1;
    *(uint4*)&lsB[arow * 64 + aseg * 8]      = pk.b0;
    *(uint4*)&lsB[arow * 64 + 32 + aseg * 8] = pk.b1;
    __syncthreads();

    // ---- prefetch next K tile (overlaps with WMMA burst) ----
    if (k0 + 64 < K)
      pk = load_tiles<GATHER>(A, Wt, hb, zbuf, pdbuf, e, e30, eg,
                              K, k0 + 64, n0, arow, aseg);

    // ---- 4 WMMAs from LDS (two 32-K chunks x two N subtiles) ----
    const int h   = lane >> 4;
    const int ml  = wm * 16 + (lane & 15);
    const int nl0 = wn * 32 + (lane & 15);
    const int nl1 = nl0 + 16;
#pragma unroll
    for (int c = 0; c < 2; ++c) {
      union { uint4 q[2]; v16bf v; } fa, fb0, fb1;
      fa.q[0]  = *(const uint4*)&lsA[ml * 64 + 32 * c + 8 * h];
      fa.q[1]  = *(const uint4*)&lsA[ml * 64 + 32 * c + 16 + 8 * h];
      fb0.q[0] = *(const uint4*)&lsB[nl0 * 64 + 32 * c + 8 * h];
      fb0.q[1] = *(const uint4*)&lsB[nl0 * 64 + 32 * c + 16 + 8 * h];
      fb1.q[0] = *(const uint4*)&lsB[nl1 * 64 + 32 * c + 8 * h];
      fb1.q[1] = *(const uint4*)&lsB[nl1 * 64 + 32 * c + 16 + 8 * h];
      acc0 = __builtin_amdgcn_wmma_f32_16x16x32_bf16(false, fa.v, false, fb0.v,
                                                     (short)0, acc0, false, false);
      acc1 = __builtin_amdgcn_wmma_f32_16x16x32_bf16(false, fa.v, false, fb1.v,
                                                     (short)0, acc1, false, false);
    }
    __syncthreads();
  }

  // ---- store (C/D layout: VGPR r -> M = r + 8*(lane>>4), N = lane&15) ----
  const int hh = lane >> 4, nn = lane & 15;
#pragma unroll
  for (int s = 0; s < 2; ++s) {
    v8f a = s ? acc1 : acc0;
    int col = n0 + wn * 32 + s * 16 + nn;
    float bs = bias ? bias[col] : 0.f;
#pragma unroll
    for (int r = 0; r < 8; ++r) {
      int row = m0 + wm * 16 + 8 * hh + r;
      float v = a[r] + bs;
      if (relu) v = fmaxf(v, 0.f);
      size_t idx = (size_t)row * Np + col;
      if (outF) outF[idx] = v;
      if (outH) outH[idx] = f2bf(v);
    }
  }
}

// ---------------------------------------------------------------------------
// Node prep: quat->R, trans, dihedrals, time fourier -> node_in (bf16, Kp=128)
// ---------------------------------------------------------------------------
__global__ void prep_nodes_kernel(const float* __restrict__ bb,
                                  const float* __restrict__ rigids,
                                  const float* __restrict__ t,
                                  float* __restrict__ R, float* __restrict__ trans,
                                  unsigned short* __restrict__ nodein, int N) {
  int n = blockIdx.x * blockDim.x + threadIdx.x;
  if (n >= N) return;

  float q0 = rigids[n*7+0], q1 = rigids[n*7+1], q2 = rigids[n*7+2], q3 = rigids[n*7+3];
  float qn = sqrtf(q0*q0 + q1*q1 + q2*q2 + q3*q3) + 1e-8f;
  float w = q0/qn, x = q1/qn, y = q2/qn, z = q3/qn;
  float* Rr = R + (size_t)n * 9;
  Rr[0] = 1.f - 2.f*(y*y + z*z); Rr[1] = 2.f*(x*y - w*z);       Rr[2] = 2.f*(x*z + w*y);
  Rr[3] = 2.f*(x*y + w*z);       Rr[4] = 1.f - 2.f*(x*x + z*z); Rr[5] = 2.f*(y*z - w*x);
  Rr[6] = 2.f*(x*z - w*y);       Rr[7] = 2.f*(y*z + w*x);       Rr[8] = 1.f - 2.f*(x*x + y*y);
  trans[n*3+0] = rigids[n*7+4]; trans[n*3+1] = rigids[n*7+5]; trans[n*3+2] = rigids[n*7+6];

  const float eps = 1e-7f;
  float dc[3], dsn[3];
  for (int kk = 0; kk < 3; ++kk) {
    int k = 3 * n + kk;
    float D = 0.f;
    if (k >= 1 && k <= 3 * N - 3) {
      int j = k - 1;
      float X[4][3];
      for (int a = 0; a < 4; ++a) {
        int idx = j + a;
        for (int c = 0; c < 3; ++c)
          X[a][c] = bb[(size_t)(idx / 3) * 12 + (idx % 3) * 3 + c];
      }
      float d0[3], d1[3], d2[3], u2[3], u1[3], u0[3], c2[3], c1[3], n2[3], n1[3];
      for (int c = 0; c < 3; ++c) { d0[c] = X[1][c]-X[0][c]; d1[c] = X[2][c]-X[1][c]; d2[c] = X[3][c]-X[2][c]; }
      unit3(d0, u2, eps); unit3(d1, u1, eps); unit3(d2, u0, eps);
      cross3(u2, u1, c2); unit3(c2, n2, eps);
      cross3(u1, u0, c1); unit3(c1, n1, eps);
      float cosD = dot3(n2, n1);
      cosD = fminf(fmaxf(cosD, -1.f + eps), 1.f - eps);
      float s = dot3(u2, n1);
      float sg = (s > 0.f) ? 1.f : ((s < 0.f) ? -1.f : 0.f);
      D = sg * acosf(cosD);
    }
    dc[kk] = cosf(D); dsn[kk] = sinf(D);
  }

  unsigned short* row = nodein + (size_t)n * 128;
  for (int kk = 0; kk < 3; ++kk) { row[kk] = f2bf(dc[kk]); row[3 + kk] = f2bf(dsn[kk]); }
  float tv = t[0];
  const float lstep = logf(1000.f) / 31.f;
  for (int i = 0; i < 32; ++i) {
    float f = expf((float)i * lstep);
    float ang = tv * f;
    row[6 + i]  = f2bf(sinf(ang));
    row[38 + i] = f2bf(cosf(ang));
  }
  for (int c = 70; c < 128; ++c) row[c] = 0;
}

// ---------------------------------------------------------------------------
// KNN: block per row, d2 row in LDS, 30 argmin passes
// ---------------------------------------------------------------------------
__global__ __launch_bounds__(256)
void knn_kernel(const float* __restrict__ xca, int* __restrict__ ei0, int N, int K) {
  __shared__ float d2s[2048];
  __shared__ float redv[256];
  __shared__ int   redi[256];
  int i = blockIdx.x, tid = threadIdx.x;
  float x0 = xca[i*3+0], x1 = xca[i*3+1], x2 = xca[i*3+2];
  for (int j = tid; j < N; j += 256) {
    float dx = x0 - xca[j*3+0], dy = x1 - xca[j*3+1], dz = x2 - xca[j*3+2];
    float d2 = dx*dx + dy*dy + dz*dz;
    if (j == i) d2 += 1e12f;
    d2s[j] = d2;
  }
  __syncthreads();
  for (int kk = 0; kk < K; ++kk) {
    float best = 3.0e38f; int bi = N;
    for (int j = tid; j < N; j += 256) {
      float v = d2s[j];
      if (v < best) { best = v; bi = j; }
    }
    redv[tid] = best; redi[tid] = bi;
    __syncthreads();
    for (int o = 128; o > 0; o >>= 1) {
      if (tid < o) {
        float ov = redv[tid + o]; int oi = redi[tid + o];
        if (ov < redv[tid] || (ov == redv[tid] && oi < redi[tid])) {
          redv[tid] = ov; redi[tid] = oi;
        }
      }
      __syncthreads();
    }
    if (tid == 0) { ei0[(size_t)i * K + kk] = redi[0]; d2s[redi[0]] = 3.4e38f; }
    __syncthreads();
  }
}

// ---------------------------------------------------------------------------
// Edge features: 16 atom-pair RBFs (256) + positional embedding (16) -> bf16
// ---------------------------------------------------------------------------
__global__ void edge_feat_kernel(const float* __restrict__ bb,
                                 const int* __restrict__ ei0,
                                 unsigned short* __restrict__ ein, int E) {
  int e = blockIdx.x * blockDim.x + threadIdx.x;
  if (e >= E) return;
  int dst = ei0[e];
  int src = e / 30;
  unsigned short* row = ein + (size_t)e * 320;
  const float sigma = 1.25f;              // (22-2)/16
  for (int a = 0; a < 4; ++a) {
    float s0 = bb[(size_t)src*12 + a*3 + 0];
    float s1 = bb[(size_t)src*12 + a*3 + 1];
    float s2 = bb[(size_t)src*12 + a*3 + 2];
    for (int c = 0; c < 4; ++c) {
      float dv0 = s0 - bb[(size_t)dst*12 + c*3 + 0] + 1e-8f;
      float dv1 = s1 - bb[(size_t)dst*12 + c*3 + 1] + 1e-8f;
      float dv2 = s2 - bb[(size_t)dst*12 + c*3 + 2] + 1e-8f;
      float dist = sqrtf(dv0*dv0 + dv1*dv1 + dv2*dv2);
      int base = (a * 4 + c) * 16;
      for (int r = 0; r < 16; ++r) {
        float mu = 2.f + (20.f / 15.f) * (float)r;
        float u = (dist - mu) / sigma;
        row[base + r] = f2bf(expf(-u * u));
      }
    }
  }
  float dd = (float)(dst - src);
  const float lf = logf(10000.f) / 16.f;
  for (int j = 0; j < 8; ++j) {
    float f = expf(-(float)(2 * j) * lf);
    float ang = dd * f;
    row[256 + j] = f2bf(cosf(ang));
    row[264 + j] = f2bf(sinf(ang));
  }
  for (int c = 272; c < 320; ++c) row[c] = 0;
}

// ---------------------------------------------------------------------------
// Fused (optional residual) + LayerNorm; wave per row; N%32==0, N<=512
// ---------------------------------------------------------------------------
__global__ __launch_bounds__(256)
void ln_rows_kernel(const float* __restrict__ x, const float* __restrict__ res,
                    const float* __restrict__ g, const float* __restrict__ b,
                    float* __restrict__ outF, unsigned short* __restrict__ outH,
                    int M, int N) {
  int row = blockIdx.x * 8 + (threadIdx.x >> 5);
  int lane = threadIdx.x & 31;
  if (row >= M) return;
  const float* xr = x + (size_t)row * N;
  const float* rr = res ? res + (size_t)row * N : nullptr;
  float v[16];
  int nper = N >> 5;
  float s = 0.f;
  for (int i = 0; i < nper; ++i) {
    int c = lane + (i << 5);
    float t = xr[c] + (rr ? rr[c] : 0.f);
    v[i] = t; s += t;
  }
  for (int o = 16; o; o >>= 1) s += __shfl_xor(s, o, 32);
  float mu = s / (float)N;
  float q = 0.f;
  for (int i = 0; i < nper; ++i) { float d = v[i] - mu; q += d * d; }
  for (int o = 16; o; o >>= 1) q += __shfl_xor(q, o, 32);
  float inv = rsqrtf(q / (float)N + 1e-5f);
  for (int i = 0; i < nper; ++i) {
    int c = lane + (i << 5);
    float o = (v[i] - mu) * inv * g[c] + b[c];
    size_t idx = (size_t)row * N + c;
    if (outF) outF[idx] = o;
    if (outH) outH[idx] = f2bf(o);
  }
}

// ---------------------------------------------------------------------------
// s_out = LN(h + segsum(m)/den); block per node, 512 channels
// ---------------------------------------------------------------------------
__global__ __launch_bounds__(256)
void s_out_kernel(const unsigned short* __restrict__ m2o,
                  const float* __restrict__ node, const float* __restrict__ lat,
                  const float* __restrict__ sg, const float* __restrict__ sb,
                  float* __restrict__ sout, unsigned short* __restrict__ soutb) {
  __shared__ float red[256];
  int n = blockIdx.x, tid = threadIdx.x;
  float vals[2];
  float s = 0.f;
  for (int ii = 0; ii < 2; ++ii) {
    int c = tid + ii * 256;
    float num = 0.f;
    const unsigned short* base = m2o + (size_t)n * 30 * 512 + c;
    for (int j = 0; j < 30; ++j) num += bf2f(base[(size_t)j * 512]);
    float h = (c < 256) ? node[(size_t)n * 256 + c]
                        : ((c < 384) ? lat[(size_t)n * 128 + (c - 256)] : 0.f);
    float t = h + num / (30.f + 1e-8f);
    vals[ii] = t; s += t;
  }
  red[tid] = s; __syncthreads();
  for (int o = 128; o; o >>= 1) { if (tid < o) red[tid] += red[tid + o]; __syncthreads(); }
  float mu = red[0] / 512.f;
  __syncthreads();
  float q = 0.f;
  for (int ii = 0; ii < 2; ++ii) { float d = vals[ii] - mu; q += d * d; }
  red[tid] = q; __syncthreads();
  for (int o = 128; o; o >>= 1) { if (tid < o) red[tid] += red[tid + o]; __syncthreads(); }
  float inv = rsqrtf(red[0] / 512.f + 1e-5f);
  for (int ii = 0; ii < 2; ++ii) {
    int c = tid + ii * 256;
    float o = (vals[ii] - mu) * inv * sg[c] + sb[c];
    sout[(size_t)n * 512 + c] = o;
    soutb[(size_t)n * 512 + c] = f2bf(o);
  }
}

// ---------------------------------------------------------------------------
// pts = R @ ptsraw + trans ; pd = ||pts[ei0]-pts[ei1]+1e-8|| ; h build ; latent
// ---------------------------------------------------------------------------
__global__ void apply_rigid_kernel(const float* __restrict__ ptsraw,
                                   const float* __restrict__ R,
                                   const float* __restrict__ trans,
                                   float* __restrict__ pts, int N) {
  int idx = blockIdx.x * blockDim.x + threadIdx.x;
  if (idx >= N * 8) return;
  int n = idx >> 3, p = idx & 7;
  const float* v = ptsraw + (size_t)n * 64 + p * 3;
  const float* Rr = R + (size_t)n * 9;
  for (int i = 0; i < 3; ++i) {
    float o = Rr[i*3+0]*v[0] + Rr[i*3+1]*v[1] + Rr[i*3+2]*v[2] + trans[n*3+i];
    pts[(size_t)n * 24 + p * 3 + i] = o;
  }
}

__global__ void pd_kernel(const float* __restrict__ pts, const int* __restrict__ ei0,
                          unsigned short* __restrict__ pdb, int E) {
  int e = blockIdx.x * blockDim.x + threadIdx.x;
  if (e >= E) return;
  int d = ei0[e], s = e / 30;
  for (int p = 0; p < 8; ++p) {
    float a0 = pts[(size_t)d*24 + p*3 + 0] - pts[(size_t)s*24 + p*3 + 0] + 1e-8f;
    float a1 = pts[(size_t)d*24 + p*3 + 1] - pts[(size_t)s*24 + p*3 + 1] + 1e-8f;
    float a2 = pts[(size_t)d*24 + p*3 + 2] - pts[(size_t)s*24 + p*3 + 2] + 1e-8f;
    pdb[(size_t)e * 64 + p] = f2bf(sqrtf(a0*a0 + a1*a1 + a2*a2));
  }
  for (int p = 8; p < 64; ++p) pdb[(size_t)e * 64 + p] = 0;
}

__global__ void h_build_kernel(const float* __restrict__ node, const float* __restrict__ lat,
                               unsigned short* __restrict__ hb, int N) {
  int idx = blockIdx.x * blockDim.x + threadIdx.x;
  if (idx >= N * 512) return;
  int n = idx >> 9, c = idx & 511;
  float v = (c < 256) ? node[(size_t)n * 256 + c]
                      : ((c < 384) ? lat[(size_t)n * 128 + (c - 256)] : 0.f);
  hb[idx] = f2bf(v);
}

__global__ void latent_update_kernel(float* __restrict__ lat,
                                     const float* __restrict__ up,
                                     const float* __restrict__ gate, int n) {
  int i = blockIdx.x * blockDim.x + threadIdx.x;
  if (i >= n) return;
  float g = gate[i];
  lat[i] += up[i] * (1.f / (1.f + expf(-g)));
}

// ---------------------------------------------------------------------------
// Host orchestration
// ---------------------------------------------------------------------------
extern "C" void kernel_launch(void* const* d_in, const int* in_sizes, int n_in,
                              void* d_out, int out_size, void* d_ws, size_t ws_size,
                              hipStream_t stream) {
  (void)in_sizes; (void)n_in; (void)out_size; (void)ws_size;
  const int N = 2048, KNN = 30, E = N * KNN;   // 61440

  auto P = [&](int i) -> const float* { return (const float*)d_in[i]; };
  // Param leaf order (jax tree / sorted-dict flattening):
  // 0 latent,1 x_ca,2 bb,3 rigids7,4 t,5 res_mask,
  // 6..13 embed_edge {b,g,l0.b,l0.w,l1.b,l1.w,l2.b,l2.w}
  // 14..21 embed_node {b,g,l0.b,l0.w,l1.b,l1.w,l2.b,l2.w}
  // per layer l at LB=22+22*l:
  //  +0 ez.b +1 ez.w +2 lg.b +3 lg.w +4 lu.b +5 lu.w +6 m0.b +7 m0.w
  //  +8 m1.b +9 m1.w +10 m2.b +11 m2.w +12 nb +13 ng +14 nu.b +15 nu.w
  //  +16 pt.b +17 pt.w +18 sb +19 sg +20 zb +21 zg

  // ----- workspace layout -----
  char* base = (char*)d_ws;
  size_t off = 0;
  auto alloc = [&](size_t bytes) -> void* {
    void* p = base + off;
    off = (off + bytes + 255) & ~(size_t)255;
    return p;
  };
  // packed weights (bf16, transposed Np x Kp)
  void* wEn0 = alloc((size_t)128 * 512 * 2);
  void* wEn1 = alloc((size_t)512 * 512 * 2);
  void* wEn2 = alloc((size_t)512 * 256 * 2);
  void* wEe0 = alloc((size_t)320 * 256 * 2);
  void* wEe1 = alloc((size_t)256 * 256 * 2);
  void* wEe2 = alloc((size_t)256 * 128 * 2);
  void *wPt[4], *wM0[4], *wM1[4], *wM2[4], *wEz[4], *wLu[4], *wLg[4], *wNu[4], *bPt[4];
  for (int l = 0; l < 4; ++l) {
    wPt[l] = alloc((size_t)512  * 64  * 2);
    wM0[l] = alloc((size_t)1216 * 256 * 2);
    wM1[l] = alloc((size_t)256  * 256 * 2);
    wM2[l] = alloc((size_t)256  * 512 * 2);
    wEz[l] = alloc((size_t)512  * 128 * 2);
    wLu[l] = alloc((size_t)512  * 128 * 2);
    wLg[l] = alloc((size_t)512  * 128 * 2);
    wNu[l] = alloc((size_t)512  * 256 * 2);
    bPt[l] = alloc((size_t)64 * 4);           // only pt bias needs padding
  }
  // activations
  float* R       = (float*)alloc((size_t)N * 9 * 4);
  float* trans   = (float*)alloc((size_t)N * 3 * 4);
  unsigned short* nodein = (unsigned short*)alloc((size_t)N * 128 * 2);
  unsigned short* nh1    = (unsigned short*)alloc((size_t)N * 512 * 2);
  unsigned short* nh2    = (unsigned short*)alloc((size_t)N * 512 * 2);
  float* noderaw = (float*)alloc((size_t)N * 256 * 4);
  float* node    = (float*)alloc((size_t)N * 256 * 4);
  int*   ei0     = (int*)alloc((size_t)E * 4);
  unsigned short* edgein = (unsigned short*)alloc((size_t)E * 320 * 2);
  unsigned short* mh1    = (unsigned short*)alloc((size_t)E * 256 * 2);
  unsigned short* mh2    = (unsigned short*)alloc((size_t)E * 256 * 2);
  float* zraw    = (float*)alloc((size_t)E * 128 * 4);
  float* z       = (float*)alloc((size_t)E * 128 * 4);
  unsigned short* zb16   = (unsigned short*)alloc((size_t)E * 128 * 2);
  unsigned short* hbf    = (unsigned short*)alloc((size_t)N * 512 * 2);
  float* ptsraw  = (float*)alloc((size_t)N * 64 * 4);
  float* pts     = (float*)alloc((size_t)N * 24 * 4);
  unsigned short* pdb    = (unsigned short*)alloc((size_t)E * 64 * 2);
  unsigned short* m2o    = (unsigned short*)alloc((size_t)E * 512 * 2);
  float* ezo     = (float*)alloc((size_t)E * 128 * 4);
  float* sout    = (float*)alloc((size_t)N * 512 * 4);
  unsigned short* soutb  = (unsigned short*)alloc((size_t)N * 512 * 2);
  float* latw    = (float*)alloc((size_t)N * 128 * 4);
  float* latup   = (float*)alloc((size_t)N * 128 * 4);
  float* latgt   = (float*)alloc((size_t)N * 128 * 4);
  float* ndup    = (float*)alloc((size_t)N * 256 * 4);

  auto packW = [&](const float* w, void* wp, int K, int Nn, int Kp, int Np) {
    int total = Kp * Np;
    pack_w_kernel<<<(total + 255) / 256, 256, 0, stream>>>(w, (unsigned short*)wp, K, Nn, Kp, Np);
  };
  auto gemm = [&](const void* A, const void* W, const void* bias,
                  void* outF, void* outH, int M, int K, int Np, int relu) {
    dim3 g(M / 64, Np / 64);
    gemm_wmma<false><<<g, 256, 0, stream>>>((const unsigned short*)A, (const unsigned short*)W,
                                            (const float*)bias, (float*)outF, (unsigned short*)outH,
                                            M, K, Np, relu, nullptr, nullptr, nullptr, nullptr);
  };
  auto gemm_gather = [&](const void* W, const void* bias, void* outH, int relu) {
    dim3 g(E / 64, 256 / 64);
    gemm_wmma<true><<<g, 256, 0, stream>>>(nullptr, (const unsigned short*)W, (const float*)bias,
                                           nullptr, (unsigned short*)outH, E, 1216, 256, relu,
                                           hbf, zb16, pdb, ei0);
  };
  auto ln = [&](const float* x, const float* res, const float* g, const float* b,
                float* oF, unsigned short* oH, int M, int Nn) {
    ln_rows_kernel<<<(M + 7) / 8, 256, 0, stream>>>(x, res, g, b, oF, oH, M, Nn);
  };

  // ----- pack all weights (transposed) -----
  packW(P(17), wEn0, 70, 512, 128, 512);
  packW(P(19), wEn1, 512, 512, 512, 512);
  packW(P(21), wEn2, 512, 256, 512, 256);
  packW(P(9),  wEe0, 272, 256, 320, 256);
  packW(P(11), wEe1, 256, 256, 256, 256);
  packW(P(13), wEe2, 256, 128, 256, 128);
  for (int l = 0; l < 4; ++l) {
    int LB = 22 + 22 * l;
    packW(P(LB + 17), wPt[l], 512, 24, 512, 64);
    packW(P(LB + 7),  wM0[l], 1160, 256, 1216, 256);
    packW(P(LB + 9),  wM1[l], 256, 256, 256, 256);
    packW(P(LB + 11), wM2[l], 256, 512, 256, 512);
    packW(P(LB + 1),  wEz[l], 512, 128, 512, 128);
    packW(P(LB + 5),  wLu[l], 512, 128, 512, 128);
    packW(P(LB + 3),  wLg[l], 512, 128, 512, 128);
    packW(P(LB + 15), wNu[l], 512, 256, 512, 256);
    pack_b_kernel<<<1, 64, 0, stream>>>(P(LB + 16), (float*)bPt[l], 24, 64);
  }

  // ----- latent copy, node prep, KNN, edge features -----
  copy_f32_kernel<<<(N * 128 + 255) / 256, 256, 0, stream>>>(P(0), latw, N * 128);
  prep_nodes_kernel<<<(N + 255) / 256, 256, 0, stream>>>(P(2), P(3), P(4), R, trans, nodein, N);
  knn_kernel<<<N, 256, 0, stream>>>(P(1), ei0, N, KNN);
  edge_feat_kernel<<<(E + 255) / 256, 256, 0, stream>>>(P(2), ei0, edgein, E);

  // ----- node embed MLP + LN -----
  gemm(nodein, wEn0, P(16), nullptr, nh1, N, 128, 512, 1);
  gemm(nh1,    wEn1, P(18), nullptr, nh2, N, 512, 512, 1);
  gemm(nh2,    wEn2, P(20), noderaw, nullptr, N, 512, 256, 0);
  ln(noderaw, nullptr, P(15), P(14), node, nullptr, N, 256);

  // ----- edge embed MLP + LN -----
  gemm(edgein, wEe0, P(8),  nullptr, mh1, E, 320, 256, 1);
  gemm(mh1,    wEe1, P(10), nullptr, mh2, E, 256, 256, 1);
  gemm(mh2,    wEe2, P(12), zraw, nullptr, E, 256, 128, 0);
  ln(zraw, nullptr, P(7), P(6), z, zb16, E, 128);

  // ----- message-passing layers -----
  for (int l = 0; l < 4; ++l) {
    int LB = 22 + 22 * l;
    h_build_kernel<<<(N * 512 + 255) / 256, 256, 0, stream>>>(node, latw, hbf, N);
    gemm(hbf, wPt[l], bPt[l], ptsraw, nullptr, N, 512, 64, 0);
    apply_rigid_kernel<<<(N * 8 + 255) / 256, 256, 0, stream>>>(ptsraw, R, trans, pts, N);
    pd_kernel<<<(E + 255) / 256, 256, 0, stream>>>(pts, ei0, pdb, E);

    gemm_gather(wM0[l], P(LB + 6), mh1, 1);
    gemm(mh1, wM1[l], P(LB + 8),  nullptr, mh2, E, 256, 256, 1);
    gemm(mh2, wM2[l], P(LB + 10), nullptr, m2o, E, 256, 512, 0);

    s_out_kernel<<<N, 256, 0, stream>>>(m2o, node, latw, P(LB + 19), P(LB + 18), sout, soutb);

    gemm(m2o, wEz[l], P(LB + 0), ezo, nullptr, E, 512, 128, 0);
    ln(z, ezo, P(LB + 21), P(LB + 20), z, zb16, E, 128);

    gemm(soutb, wLu[l], P(LB + 4), latup, nullptr, N, 512, 128, 0);
    gemm(soutb, wLg[l], P(LB + 2), latgt, nullptr, N, 512, 128, 0);
    latent_update_kernel<<<(N * 128 + 255) / 256, 256, 0, stream>>>(latw, latup, latgt, N * 128);

    gemm(soutb, wNu[l], P(LB + 14), ndup, nullptr, N, 512, 256, 0);
    ln(node, ndup, P(LB + 13), P(LB + 12), node, nullptr, N, 256);
  }

  copy_f32_kernel<<<(N * 128 + 255) / 256, 256, 0, stream>>>(latw, (float*)d_out, N * 128);
}
